// PCUpSample_53017076302429
// MI455X (gfx1250) — matrix operand order repaired and checked by reading, compiled, not run on hardware
//
#include <hip/hip_runtime.h>

typedef __attribute__((ext_vector_type(16))) _Float16 v16h;
typedef __attribute__((ext_vector_type(8)))  _Float16 v8h;
typedef __attribute__((ext_vector_type(8)))  float    v8f;
typedef int v4i __attribute__((vector_size(4 * sizeof(int))));
typedef __attribute__((address_space(1))) v4i* gv4i_ptr;
typedef __attribute__((address_space(3))) v4i* lv4i_ptr;

#define B_    4
#define N_    16384
#define M_    4096
#define C1_   128
#define C2_   256
#define CIN_  384      // C1 + C2
#define H_    512
#define COUT_ 256
#define KNN_  16
#define G_    (B_ * N_)   // 65536 columns

// ---------------------------------------------------------------------------
// KNN: for each (b,n), 16 nearest of M points. xyz2 (48 KB) staged in LDS via
// CDNA5 async global->LDS (ASYNCcnt-tracked), then distances computed from LDS.
// ---------------------------------------------------------------------------
__global__ void knn_kernel(const float* __restrict__ xyz1,
                           const float* __restrict__ xyz2,
                           int* __restrict__ idx, float* __restrict__ wt) {
    __shared__ __align__(16) float sxyz[3 * M_];   // [3][M] same layout as global
    int b = blockIdx.y;
    const float* p2 = xyz2 + (size_t)b * 3 * M_;

    // 3*M floats = 3072 x 16B chunks, async-copied to LDS
    for (int i = threadIdx.x; i < (3 * M_) / 4; i += blockDim.x) {
        __builtin_amdgcn_global_load_async_to_lds_b128(
            (gv4i_ptr)(p2 + 4 * i),
            (lv4i_ptr)(&sxyz[4 * i]),
            0, 0);
    }
    asm volatile("s_wait_asynccnt 0" ::: "memory");
    __syncthreads();

    int n = blockIdx.x * blockDim.x + threadIdx.x;
    const float* p1 = xyz1 + (size_t)b * 3 * N_;
    float px = p1[n], py = p1[N_ + n], pz = p1[2 * N_ + n];

    float bd[KNN_];
    int   bi[KNN_];
#pragma unroll
    for (int k = 0; k < KNN_; ++k) { bd[k] = 3.4e38f; bi[k] = 0; }

    for (int m = 0; m < M_; ++m) {
        float dx = px - sxyz[m];
        float dy = py - sxyz[M_ + m];
        float dz = pz - sxyz[2 * M_ + m];
        float d = dx * dx + dy * dy + dz * dz;
        if (d < bd[KNN_ - 1]) {
            bd[KNN_ - 1] = d; bi[KNN_ - 1] = m;
            // bubble new element into sorted position (fully unrolled: stays in VGPRs)
#pragma unroll
            for (int j = KNN_ - 1; j > 0; --j) {
                if (bd[j] < bd[j - 1]) {
                    float td = bd[j]; bd[j] = bd[j - 1]; bd[j - 1] = td;
                    int   ti = bi[j]; bi[j] = bi[j - 1]; bi[j - 1] = ti;
                }
            }
        }
    }

    float rc[KNN_];
    float rsum = 0.f;
#pragma unroll
    for (int k = 0; k < KNN_; ++k) { rc[k] = 1.0f / (bd[k] + 1e-8f); rsum += rc[k]; }
    float inv = 1.0f / rsum;
    size_t base = ((size_t)b * N_ + n) * KNN_;
#pragma unroll
    for (int k = 0; k < KNN_; ++k) { idx[base + k] = bi[k]; wt[base + k] = rc[k] * inv; }
}

// ---------------------------------------------------------------------------
// Interpolate features2 -> xT[g][C1..C1+C2) (f16). One wave per point g.
// ---------------------------------------------------------------------------
__global__ void interp_kernel(const float* __restrict__ f2,
                              const int* __restrict__ idx,
                              const float* __restrict__ wt,
                              _Float16* __restrict__ xT) {
    int warp = threadIdx.x >> 5;
    int lane = threadIdx.x & 31;
    int g = blockIdx.x * 8 + warp;
    int b = g / N_;
    const float* fb = f2 + (size_t)b * C2_ * M_;

    float acc[8] = {0.f, 0.f, 0.f, 0.f, 0.f, 0.f, 0.f, 0.f};
    size_t kb = (size_t)g * KNN_;
    for (int k = 0; k < KNN_; ++k) {
        int   m = idx[kb + k];
        float w = wt[kb + k];
#pragma unroll
        for (int j = 0; j < 8; ++j) {
            int c = lane + 32 * j;
            acc[j] += w * fb[(size_t)c * M_ + m];
        }
    }
    _Float16* out = xT + (size_t)g * CIN_ + C1_;
#pragma unroll
    for (int j = 0; j < 8; ++j) out[lane + 32 * j] = (_Float16)acc[j];
}

// features1 [B][C1][N] -> xT[g][0..C1) (f16, transposed pack)
__global__ void pack_f1_kernel(const float* __restrict__ f1, _Float16* __restrict__ xT) {
    size_t t = (size_t)blockIdx.x * blockDim.x + threadIdx.x;  // over G*C1
    int ci = (int)(t % C1_);
    size_t g = t / C1_;
    int b = (int)(g / N_);
    int n = (int)(g % N_);
    xT[g * CIN_ + ci] = (_Float16)f1[((size_t)b * C1_ + ci) * N_ + n];
}

__global__ void f32_to_f16_kernel(const float* __restrict__ in, _Float16* __restrict__ out, int n) {
    int t = blockIdx.x * blockDim.x + threadIdx.x;
    if (t < n) out[t] = (_Float16)in[t];
}

// ---------------------------------------------------------------------------
// WMMA GEMM: Y[CO][G] = A[CO][CI] * BT[G][CI]^T   (f16 in, f32 out)
// 8 waves/block, each wave computes a 16(co) x 64(col) strip: one A fragment
// reused by 4 WMMAs (4 accumulators). Block macro-tile: 32co x 256col.
// Fragment loads follow the CDNA5 f16 A/B per-lane VGPR layouts as contiguous
// v8h (16B) chunks.
// ---------------------------------------------------------------------------
__global__ __launch_bounds__(256) void wmma_gemm_kernel(
    const _Float16* __restrict__ A,    // [CO][CI] row-major (weights)
    const _Float16* __restrict__ BT,   // [G][CI]  row-major (activations, transposed)
    float* __restrict__ Y,             // [CO][G]
    int CO, int CI, int G) {
    int wave  = threadIdx.x >> 5;
    int lane  = threadIdx.x & 31;
    int lhalf = lane >> 4;     // 0: lanes 0-15, 1: lanes 16-31
    int l15   = lane & 15;

    int co_base  = blockIdx.y * 32 + (wave & 1) * 16;
    int col_base = blockIdx.x * 256 + (wave >> 1) * 64;

    const _Float16* arow  = A  + (size_t)(co_base + l15)  * CI;  // A row M = lane&15
    const _Float16* brow0 = BT + (size_t)(col_base + l15) * CI;  // B col N = lane&15

    v8f acc[4] = {};
    for (int k0 = 0; k0 < CI; k0 += 32) {
        union { v16h v; v8h h[2]; } a;
        // A 16x32 f16: halves 0-7 -> K = k0 + 8*lhalf + 0..7 ; halves 8-15 -> +16
        a.h[0] = *(const v8h*)(arow + k0 + lhalf * 8);
        a.h[1] = *(const v8h*)(arow + k0 + 16 + lhalf * 8);
        __builtin_prefetch(arow + k0 + 32, 0, 0);   // global_prefetch_b8 on next A chunk
#pragma unroll
        for (int j = 0; j < 4; ++j) {
            union { v16h v; v8h h[2]; } bm;
            // B 32x16 f16: halves 0-7 -> K = k0 + 8*lhalf + 0..7 ; halves 8-15 -> +16
            const _Float16* bp = brow0 + (size_t)j * 16 * CI + k0 + lhalf * 8;
            bm.h[0] = *(const v8h*)(bp);
            bm.h[1] = *(const v8h*)(bp + 16);
            acc[j] = __builtin_amdgcn_wmma_f32_16x16x32_f16(false, a.v, false, bm.v,
                                                            (short)0, acc[j], false, false);
        }
    }
    // C/D layout: reg r -> M = r + 8*lhalf, N = lane&15
#pragma unroll
    for (int j = 0; j < 4; ++j) {
        int g = col_base + j * 16 + l15;
#pragma unroll
        for (int r = 0; r < 8; ++r) {
            int m = co_base + r + 8 * lhalf;
            Y[(size_t)m * G + g] = acc[j][r];
        }
    }
}

// ---------------------------------------------------------------------------
// Per-channel batch stats: st[2c] = mean, st[2c+1] = rsqrt(var + eps)
// ---------------------------------------------------------------------------
__global__ void stats_kernel(const float* __restrict__ Y, float* __restrict__ st, int G) {
    int co = blockIdx.x;
    const float* row = Y + (size_t)co * G;
    float s = 0.f, s2 = 0.f;
    for (int i = threadIdx.x; i < G; i += blockDim.x) {
        float v = row[i];
        s += v; s2 += v * v;
    }
    __shared__ float sh[256], sh2[256];
    int tid = threadIdx.x;
    sh[tid] = s; sh2[tid] = s2;
    __syncthreads();
    for (int o = 128; o > 0; o >>= 1) {
        if (tid < o) { sh[tid] += sh[tid + o]; sh2[tid] += sh2[tid + o]; }
        __syncthreads();
    }
    if (tid == 0) {
        float inv = 1.0f / (float)G;
        float mu  = sh[0] * inv;
        float var = sh2[0] * inv - mu * mu;   // biased var (matches jnp.var)
        st[2 * co]     = mu;
        st[2 * co + 1] = rsqrtf(var + 1e-5f);
    }
}

// BN + ReLU, repack transposed to f16 for the next GEMM's B operand
__global__ void bn_relu_f16_kernel(const float* __restrict__ Y, const float* __restrict__ st,
                                   const float* __restrict__ gamma, const float* __restrict__ beta,
                                   _Float16* __restrict__ HT, int CO, int G) {
    size_t t = (size_t)blockIdx.x * blockDim.x + threadIdx.x;  // over CO*G
    int co = (int)(t / (size_t)G);
    size_t g = t % (size_t)G;
    float v = Y[t];
    float o = (v - st[2 * co]) * st[2 * co + 1] * gamma[co] + beta[co];
    o = fmaxf(o, 0.f);
    HT[g * (size_t)CO + co] = (_Float16)o;
}

// Final BN + ReLU, write f32 output in [B][Cout][N] layout
__global__ void bn_relu_out_kernel(const float* __restrict__ Z, const float* __restrict__ st,
                                   const float* __restrict__ gamma, const float* __restrict__ beta,
                                   float* __restrict__ out, int CO, int G, int N) {
    size_t t = (size_t)blockIdx.x * blockDim.x + threadIdx.x;  // over CO*G
    int co = (int)(t / (size_t)G);
    size_t g = t % (size_t)G;
    int b = (int)(g / (size_t)N);
    int n = (int)(g % (size_t)N);
    float v = Z[t];
    float o = fmaxf((v - st[2 * co]) * st[2 * co + 1] * gamma[co] + beta[co], 0.f);
    out[((size_t)b * CO + co) * (size_t)N + n] = o;
}

// ---------------------------------------------------------------------------
extern "C" void kernel_launch(void* const* d_in, const int* in_sizes, int n_in,
                              void* d_out, int out_size, void* d_ws, size_t ws_size,
                              hipStream_t stream) {
    const float* xyz1 = (const float*)d_in[0];
    const float* xyz2 = (const float*)d_in[1];
    const float* f1   = (const float*)d_in[2];
    const float* f2   = (const float*)d_in[3];
    const float* w0   = (const float*)d_in[4];
    // d_in[5] = b0: conv bias is cancelled exactly by the following BatchNorm -> skipped
    const float* g0   = (const float*)d_in[6];
    const float* be0  = (const float*)d_in[7];
    const float* w1   = (const float*)d_in[8];
    // d_in[9] = b1: skipped (same reason)
    const float* g1   = (const float*)d_in[10];
    const float* be1  = (const float*)d_in[11];
    float* out = (float*)d_out;

    // workspace layout (all chunks 256B-aligned sizes)
    char* ws = (char*)d_ws;
    size_t off = 0;
    int*      idx = (int*)(ws + off);      off += (size_t)G_ * KNN_ * sizeof(int);
    float*    wt  = (float*)(ws + off);    off += (size_t)G_ * KNN_ * sizeof(float);
    _Float16* w0h = (_Float16*)(ws + off); off += (size_t)H_ * CIN_ * sizeof(_Float16);
    _Float16* w1h = (_Float16*)(ws + off); off += (size_t)COUT_ * H_ * sizeof(_Float16);
    _Float16* xT  = (_Float16*)(ws + off); off += (size_t)G_ * CIN_ * sizeof(_Float16);
    _Float16* hT  = (_Float16*)(ws + off); off += (size_t)G_ * H_ * sizeof(_Float16);
    float*    y   = (float*)(ws + off);    off += (size_t)H_ * G_ * sizeof(float);
    float*    st0 = (float*)(ws + off);    off += 2 * H_ * sizeof(float);
    float*    st1 = (float*)(ws + off);    off += 2 * COUT_ * sizeof(float);
    float*    z   = y;  // y no longer needed once hT is built -> reuse for GEMM2 output

    // 1) KNN + inverse-distance weights
    knn_kernel<<<dim3(N_ / 128, B_), 128, 0, stream>>>(xyz1, xyz2, idx, wt);

    // 2) Build concatenated input xT [G][Cin] in f16
    interp_kernel<<<G_ / 8, 256, 0, stream>>>(f2, idx, wt, xT);
    pack_f1_kernel<<<(unsigned)((size_t)G_ * C1_ / 256), 256, 0, stream>>>(f1, xT);

    // 3) Weights to f16
    f32_to_f16_kernel<<<(H_ * CIN_ + 255) / 256, 256, 0, stream>>>(w0, w0h, H_ * CIN_);
    f32_to_f16_kernel<<<(COUT_ * H_ + 255) / 256, 256, 0, stream>>>(w1, w1h, COUT_ * H_);

    // 4) conv1 (WMMA GEMM) -> BN stats -> BN+ReLU repack f16
    wmma_gemm_kernel<<<dim3(G_ / 256, H_ / 32), 256, 0, stream>>>(w0h, xT, y, H_, CIN_, G_);
    stats_kernel<<<H_, 256, 0, stream>>>(y, st0, G_);
    bn_relu_f16_kernel<<<(unsigned)((size_t)H_ * G_ / 256), 256, 0, stream>>>(
        y, st0, g0, be0, hT, H_, G_);

    // 5) conv2 (WMMA GEMM) -> BN stats -> BN+ReLU -> output [B][Cout][N]
    wmma_gemm_kernel<<<dim3(G_ / 256, COUT_ / 32), 256, 0, stream>>>(w1h, hT, z, COUT_, H_, G_);
    stats_kernel<<<COUT_, 256, 0, stream>>>(z, st1, G_);
    bn_relu_out_kernel<<<(unsigned)((size_t)COUT_ * G_ / 256), 256, 0, stream>>>(
        z, st1, g1, be1, out, COUT_, G_, N_);
}
